// DeBERTaV3Classifier_68521908240798
// MI455X (gfx1250) — matrix operand order repaired and checked by reading, not compiled, not run on hardware
//
#include <hip/hip_runtime.h>
#include <hip/hip_bf16.h>
#include <math.h>

// ---------------------------------------------------------------------------
// DeBERTa-v3-base forward on gfx1250 (MI455X).
//  - All GEMMs on v_wmma_f32_16x16x32_f16 (f16 in, f32 accumulate)
//  - Large GEMMs: 64x64 block tile staged through LDS with CDNA5 async
//    global->LDS copies (ASYNCcnt) when available; 2x2 wmma tiles per wave.
//  - LN / softmax / pooling / classifier head in f32.
// ---------------------------------------------------------------------------

#define BB   4
#define SS   512
#define HH   768
#define LL   12
#define NHH  12
#define DHH  64
#define FFF  3072
#define MM   (BB * SS)      // 2048 tokens
#define SPAN2 512           // 2 * att_span
#define EPS_DB  1e-7f
#define EPS_CLS 1e-5f

typedef __attribute__((ext_vector_type(16))) _Float16 v16h;
typedef __attribute__((ext_vector_type(8)))  _Float16 v8h;
typedef __attribute__((ext_vector_type(8)))  float    v8f;
typedef __attribute__((ext_vector_type(4)))  int      v4i;

// --------------------------- async copy helpers ----------------------------
// CDNA5 GLOBAL_LOAD_ASYNC_TO_LDS_B128: LDS[dst] = MEM[src], tracked by
// ASYNCcnt. Builtin signature (from clang diagnostic): param 1 is
// 'int __vector(4) __device__ *' (global-AS v4i pointer). Guarded by
// __has_builtin; fallback = synchronous reg-staged copy.
#if __has_builtin(__builtin_amdgcn_global_load_async_to_lds_b128)
#define HAVE_ASYNC_LDS 1
#else
#define HAVE_ASYNC_LDS 0
#endif

typedef __attribute__((address_space(1))) v4i glb_v4i;
typedef __attribute__((address_space(3))) v4i lds_v4i;

__device__ __forceinline__ void async_cp16(void* lds, const void* g)
{
#if HAVE_ASYNC_LDS
    __builtin_amdgcn_global_load_async_to_lds_b128(
        (glb_v4i*)g, (lds_v4i*)lds, 0, 0);
#else
    *(v8h*)lds = *(const v8h*)g;
#endif
}

__device__ __forceinline__ void wait_async()
{
#if __has_builtin(__builtin_amdgcn_s_wait_asynccnt)
    __builtin_amdgcn_s_wait_asynccnt(0);
#else
    asm volatile("s_wait_asynccnt 0x0" ::: "memory");
#endif
}

// ---------------------------------------------------------------------------
// Direct-global batched WMMA GEMM: C[z] = A[z] * Bt[z]^T (+bias, +gelu)
//   A  : [M,K] f16 row-major, lda ;  Bt : [N,K] f16 row-major, ldb
//   per-z offsets: off = (z/zInner)*sOuter + (z%zInner)*sInner (elements)
// One wave = one 16x16 tile; block = 128 threads = 4 n-tiles.
// Used for the per-head attention GEMMs (small K, batched z=48).
// ---------------------------------------------------------------------------
__global__ __launch_bounds__(128)
void gemm_wmma(const _Float16* __restrict__ A, int lda, long sAo, long sAi,
               const _Float16* __restrict__ Bt, int ldb, long sBo, long sBi,
               const float* __restrict__ bias,
               float* __restrict__ Cf, _Float16* __restrict__ Ch,
               int ldc, long sCo, long sCi,
               int M, int N, int K, int zInner, int doGelu)
{
    int z  = blockIdx.z;
    int zo = z / zInner, zi = z % zInner;
    A  += (long)zo * sAo + (long)zi * sAi;
    Bt += (long)zo * sBo + (long)zi * sBi;
    long cOff = (long)zo * sCo + (long)zi * sCi;

    int wave = threadIdx.x >> 5;
    int lane = threadIdx.x & 31;
    int n0 = (blockIdx.x * 4 + wave) * 16;
    int m0 = blockIdx.y * 16;
    if (n0 >= N || m0 >= M) return;          // wave-uniform

    int l15 = lane & 15;
    int hi  = lane >> 4;

    const _Float16* aRow = A  + (long)(m0 + l15) * lda;
    const _Float16* bRow = Bt + (long)(n0 + l15) * ldb + (hi << 4);

    v8f acc = {};
    for (int k = 0; k < K; k += 32) {
        // A k order used here (B matches): lanes 0-15 hold K {k..k+15},
        //                     lanes 16-31 hold K {k+16..k+31}
        union { v16h v; v8h h[2]; } au;
        au.h[0] = *(const v8h*)(aRow + k + (hi << 4));
        au.h[1] = *(const v8h*)(aRow + k + 8 + (hi << 4));
        v16h bf = *(const v16h*)(bRow + k);
        acc = __builtin_amdgcn_wmma_f32_16x16x32_f16(
                  false, au.v, false, bf, (short)0, acc, false, false);
    }

    int col   = n0 + l15;
    int rbase = m0 + (hi << 3);
    float bv  = bias ? bias[col] : 0.0f;
#pragma unroll
    for (int r = 0; r < 8; ++r) {
        float v = acc[r] + bv;
        if (doGelu) v = 0.5f * v * (1.0f + erff(v * 0.70710678118654752f));
        long o = cOff + (long)(rbase + r) * ldc + col;
        if (Cf) Cf[o] = v;
        if (Ch) Ch[o] = (_Float16)v;
    }
}

// ---------------------------------------------------------------------------
// LDS-staged WMMA GEMM for the big linear layers. Block tile = 64(M) x 64(N),
// 128 threads = 4 waves in a 2x2 arrangement; each wave owns a 32x32 tile
// (2x2 wmma accumulators). Per 32-K step the block stages A(64x32 f16) and
// B(64x32 f16) into LDS via async b128 copies (4 per thread), waits on
// ASYNCcnt, barriers, then computes from LDS. M, N must be multiples of 64.
// ---------------------------------------------------------------------------
__global__ __launch_bounds__(128)
void gemm_wmma_lds(const _Float16* __restrict__ A, int lda,
                   const _Float16* __restrict__ Bt, int ldb,
                   const float* __restrict__ bias,
                   float* __restrict__ Cf, _Float16* __restrict__ Ch,
                   int ldc, int M, int N, int K, int doGelu)
{
    __shared__ __align__(64) _Float16 As[64 * 32];
    __shared__ __align__(64) _Float16 Bs[64 * 32];

    int t    = threadIdx.x;
    int lane = t & 31;
    int wave = t >> 5;
    int wm   = wave >> 1;              // 0..1
    int wn   = wave & 1;               // 0..1
    int mBlk = blockIdx.y * 64;
    int nBlk = blockIdx.x * 64;

    // staging: 64 rows x 4 b128-chunks = 256 chunks per tile; 2 per thread
    int c0   = t * 2;
    int row0 = c0 >> 2,        col0 = (c0 & 3) * 8;        // halves
    int row1 = (c0 + 1) >> 2,  col1 = ((c0 + 1) & 3) * 8;

    int l15 = lane & 15;
    int hi  = lane >> 4;

    v8f acc00 = {}, acc01 = {}, acc10 = {}, acc11 = {};

    const _Float16* aBase = A  + (long)mBlk * lda;
    const _Float16* bBase = Bt + (long)nBlk * ldb;

    for (int k = 0; k < K; k += 32) {
        __syncthreads();   // previous iteration's LDS reads complete
        async_cp16(&As[row0 * 32 + col0], aBase + (long)row0 * lda + k + col0);
        async_cp16(&As[row1 * 32 + col1], aBase + (long)row1 * lda + k + col1);
        async_cp16(&Bs[row0 * 32 + col0], bBase + (long)row0 * ldb + k + col0);
        async_cp16(&Bs[row1 * 32 + col1], bBase + (long)row1 * ldb + k + col1);
        wait_async();
        __syncthreads();

        // A fragments (m-subtiles 0,1 of this wave's 32 rows)
        const _Float16* ar0 = As + (wm * 32 + l15) * 32;
        const _Float16* ar1 = ar0 + 16 * 32;
        union { v16h v; v8h h[2]; } a0, a1;
        a0.h[0] = *(const v8h*)(ar0 + (hi << 4));
        a0.h[1] = *(const v8h*)(ar0 + 8 + (hi << 4));
        a1.h[0] = *(const v8h*)(ar1 + (hi << 4));
        a1.h[1] = *(const v8h*)(ar1 + 8 + (hi << 4));
        // B fragments (n-subtiles 0,1)
        const _Float16* br0 = Bs + (wn * 32 + l15) * 32 + (hi << 4);
        v16h b0 = *(const v16h*)(br0);
        v16h b1 = *(const v16h*)(br0 + 16 * 32);

        acc00 = __builtin_amdgcn_wmma_f32_16x16x32_f16(false, a0.v, false, b0, (short)0, acc00, false, false);
        acc01 = __builtin_amdgcn_wmma_f32_16x16x32_f16(false, a0.v, false, b1, (short)0, acc01, false, false);
        acc10 = __builtin_amdgcn_wmma_f32_16x16x32_f16(false, a1.v, false, b0, (short)0, acc10, false, false);
        acc11 = __builtin_amdgcn_wmma_f32_16x16x32_f16(false, a1.v, false, b1, (short)0, acc11, false, false);
    }

    // epilogue: 4 tiles of 16x16
    auto store_tile = [&](const v8f& acc, int mT, int nT) {
        int col   = nBlk + wn * 32 + nT * 16 + l15;
        int rbase = mBlk + wm * 32 + mT * 16 + (hi << 3);
        float bv  = bias ? bias[col] : 0.0f;
#pragma unroll
        for (int r = 0; r < 8; ++r) {
            float v = acc[r] + bv;
            if (doGelu) v = 0.5f * v * (1.0f + erff(v * 0.70710678118654752f));
            long o = (long)(rbase + r) * ldc + col;
            if (Cf) Cf[o] = v;
            if (Ch) Ch[o] = (_Float16)v;
        }
    };
    store_tile(acc00, 0, 0);
    store_tile(acc01, 0, 1);
    store_tile(acc10, 1, 0);
    store_tile(acc11, 1, 1);
}

// ---------------------------------------------------------------------------
// Weight prep: W f32 [K][N] -> Wt f16 [N][K]
// ---------------------------------------------------------------------------
__global__ void transpose_w(const float* __restrict__ W, _Float16* __restrict__ Wt,
                            int K, int N)
{
    long idx = (long)blockIdx.x * blockDim.x + threadIdx.x;
    if (idx >= (long)K * N) return;
    int  n = (int)(idx % N);
    long k = idx / N;
    Wt[(long)n * K + k] = (_Float16)W[idx];
}

// vt[b,h,d,j] = v16[(b*S+j)*H + h*DH + d]
__global__ void transpose_v(const _Float16* __restrict__ v, _Float16* __restrict__ vt)
{
    long idx = (long)blockIdx.x * blockDim.x + threadIdx.x;
    if (idx >= (long)BB * NHH * DHH * SS) return;
    int j = (int)(idx % SS); long r = idx / SS;
    int d = (int)(r % DHH);  r /= DHH;
    int h = (int)(r % NHH);
    int b = (int)(r / NHH);
    vt[idx] = v[((long)(b * SS + j)) * HH + h * DHH + d];
}

// ---------------------------------------------------------------------------
// HF make_log_bucket_position (float64 math to match numpy), clipped tables.
// ---------------------------------------------------------------------------
__global__ void bucket_kernel(int* __restrict__ c2p, int* __restrict__ p2c)
{
    int idx = blockIdx.x * blockDim.x + threadIdx.x;
    if (idx >= SS * SS) return;
    int i = idx / SS, j = idx % SS;
    int rel = i - j;
    const int mid = 128;               // BUCKETS/2
    int sgn = (rel > 0) - (rel < 0);
    int abs_pos = (rel < mid && rel > -mid) ? (mid - 1) : (rel < 0 ? -rel : rel);
    double log_pos = ceil(log((double)abs_pos / (double)mid) /
                          log(511.0 / (double)mid) * (double)(mid - 1)) + (double)mid;
    int bucket = (abs_pos <= mid) ? rel : (int)(log_pos * (double)sgn);
    int c = bucket + 256; c = c < 0 ? 0 : (c > 511 ? 511 : c);
    int p = -bucket + 256; p = p < 0 ? 0 : (p > 511 ? 511 : p);
    c2p[idx] = c;
    p2c[idx] = p;
}

// ---------------------------------------------------------------------------
// LayerNorm over 768 cols (3 elems/thread, 256 threads). Optional residual.
// ---------------------------------------------------------------------------
__global__ __launch_bounds__(256)
void ln768(const float* __restrict__ x, const float* __restrict__ add,
           const float* __restrict__ w, const float* __restrict__ b, float eps,
           float* __restrict__ out32, _Float16* __restrict__ out16)
{
    int row = blockIdx.x, t = threadIdx.x;
    __shared__ float red[256];
    float v[3]; float s = 0.f;
    long base = (long)row * HH;
#pragma unroll
    for (int e = 0; e < 3; ++e) {
        int c = t + e * 256;
        float xv = x[base + c];
        if (add) xv += add[base + c];
        v[e] = xv; s += xv;
    }
    red[t] = s; __syncthreads();
    for (int o = 128; o > 0; o >>= 1) { if (t < o) red[t] += red[t + o]; __syncthreads(); }
    float mean = red[0] * (1.0f / HH); __syncthreads();
    s = 0.f;
#pragma unroll
    for (int e = 0; e < 3; ++e) { float d = v[e] - mean; s += d * d; }
    red[t] = s; __syncthreads();
    for (int o = 128; o > 0; o >>= 1) { if (t < o) red[t] += red[t + o]; __syncthreads(); }
    float inv = 1.0f / sqrtf(red[0] * (1.0f / HH) + eps);
#pragma unroll
    for (int e = 0; e < 3; ++e) {
        int c = t + e * 256;
        float y = (v[e] - mean) * inv * w[c] + b[c];
        if (out32) out32[base + c] = y;
        if (out16) out16[base + c] = (_Float16)y;
    }
}

// Embedding gather + LayerNorm (eps = 1e-7)
__global__ __launch_bounds__(256)
void embed_ln(const int* __restrict__ ids, const float* __restrict__ emb,
              const float* __restrict__ w, const float* __restrict__ b,
              float* __restrict__ h32, _Float16* __restrict__ h16)
{
    int row = blockIdx.x, t = threadIdx.x;
    __shared__ float red[256];
    long src = (long)ids[row] * HH;
    long dst = (long)row * HH;
    float v[3]; float s = 0.f;
#pragma unroll
    for (int e = 0; e < 3; ++e) { v[e] = emb[src + t + e * 256]; s += v[e]; }
    red[t] = s; __syncthreads();
    for (int o = 128; o > 0; o >>= 1) { if (t < o) red[t] += red[t + o]; __syncthreads(); }
    float mean = red[0] * (1.0f / HH); __syncthreads();
    s = 0.f;
#pragma unroll
    for (int e = 0; e < 3; ++e) { float d = v[e] - mean; s += d * d; }
    red[t] = s; __syncthreads();
    for (int o = 128; o > 0; o >>= 1) { if (t < o) red[t] += red[t + o]; __syncthreads(); }
    float inv = 1.0f / sqrtf(red[0] * (1.0f / HH) + EPS_DB);
#pragma unroll
    for (int e = 0; e < 3; ++e) {
        int c = t + e * 256;
        float y = (v[e] - mean) * inv * w[c] + b[c];
        h32[dst + c] = y; h16[dst + c] = (_Float16)y;
    }
}

// ---------------------------------------------------------------------------
// scores[i,j] = (c2c[i,j] + cp[i, c2p_idx[i,j]] + pc[j, p2c_idx[j,i]]) * scale
// mask, row softmax, f16 probs.  grid=(S, B*NH), block=512 (1 col/thread)
// ---------------------------------------------------------------------------
__global__ __launch_bounds__(512)
void combine_softmax(const float* __restrict__ scores, const float* __restrict__ cp,
                     const float* __restrict__ pc,
                     const int* __restrict__ c2p_idx, const int* __restrict__ p2c_idx,
                     const int* __restrict__ amask,
                     _Float16* __restrict__ probs, float scale)
{
    int i = blockIdx.x;
    int z = blockIdx.y;
    int b = z / NHH;
    int j = threadIdx.x;
    __shared__ float red[512];

    long zi = ((long)z * SS + i) * SS;
    float s = scores[zi + j]
            + cp[zi + c2p_idx[i * SS + j]]
            + pc[((long)z * SS + j) * SS + p2c_idx[j * SS + i]];
    s *= scale;
    if (amask[b * SS + j] == 0) s = -3.4028235e38f;

    red[j] = s; __syncthreads();
    for (int o = 256; o > 0; o >>= 1) { if (j < o) red[j] = fmaxf(red[j], red[j + o]); __syncthreads(); }
    float mx = red[0]; __syncthreads();
    float e = expf(s - mx);
    red[j] = e; __syncthreads();
    for (int o = 256; o > 0; o >>= 1) { if (j < o) red[j] += red[j + o]; __syncthreads(); }
    probs[zi + j] = (_Float16)(e / red[0]);
}

// masked mean pool: pooled[b,d]
__global__ __launch_bounds__(768)
void pool_kernel(const float* __restrict__ h, const int* __restrict__ mask,
                 float* __restrict__ pooled)
{
    int b = blockIdx.x, d = threadIdx.x;
    float s = 0.f, ms = 0.f;
    for (int j = 0; j < SS; ++j) {
        float m = (float)mask[b * SS + j];
        s  += h[((long)(b * SS + j)) * HH + d] * m;
        ms += m;
    }
    pooled[b * HH + d] = s / fmaxf(ms, 1e-9f);
}

// tiny f32 dense: C[m,n] = A[m,:]·W[:,n] + bias[n] (+res[m,n])
__global__ void dense_small(const float* __restrict__ A, const float* __restrict__ W,
                            const float* __restrict__ bias, const float* __restrict__ res,
                            float* __restrict__ C, int M, int N, int K)
{
    int idx = blockIdx.x * blockDim.x + threadIdx.x;
    if (idx >= M * N) return;
    int m = idx / N, n = idx % N;
    float s = bias ? bias[n] : 0.f;
    for (int k = 0; k < K; ++k) s += A[(long)m * K + k] * W[(long)k * N + n];
    if (res) s += res[idx];
    C[idx] = s;
}

// row LN (eps) then exact gelu, in-place; block=128
__global__ __launch_bounds__(128)
void ln_gelu_small(float* __restrict__ x, const float* __restrict__ w,
                   const float* __restrict__ b, int N, float eps)
{
    int row = blockIdx.x, t = threadIdx.x;
    __shared__ float red[128];
    long base = (long)row * N;
    float s = 0.f;
    for (int c = t; c < N; c += 128) s += x[base + c];
    red[t] = s; __syncthreads();
    for (int o = 64; o > 0; o >>= 1) { if (t < o) red[t] += red[t + o]; __syncthreads(); }
    float mean = red[0] / N; __syncthreads();
    s = 0.f;
    for (int c = t; c < N; c += 128) { float d = x[base + c] - mean; s += d * d; }
    red[t] = s; __syncthreads();
    for (int o = 64; o > 0; o >>= 1) { if (t < o) red[t] += red[t + o]; __syncthreads(); }
    float inv = 1.0f / sqrtf(red[0] / N + eps);
    __syncthreads();
    for (int c = t; c < N; c += 128) {
        float y = (x[base + c] - mean) * inv * w[c] + b[c];
        x[base + c] = 0.5f * y * (1.0f + erff(y * 0.70710678118654752f));
    }
}

// ---------------------------------------------------------------------------
// Host-side orchestration
// ---------------------------------------------------------------------------
static inline void launch_gemm(hipStream_t st,
    const _Float16* A, int lda, long sAo, long sAi,
    const _Float16* Bt, int ldb, long sBo, long sBi,
    const float* bias, float* Cf, _Float16* Ch, int ldc, long sCo, long sCi,
    int M, int N, int K, int Z, int zInner, int gelu)
{
    dim3 grid(N / 64, M / 16, Z);
    gemm_wmma<<<grid, 128, 0, st>>>(A, lda, sAo, sAi, Bt, ldb, sBo, sBi,
                                    bias, Cf, Ch, ldc, sCo, sCi,
                                    M, N, K, zInner, gelu);
}

static inline void launch_gemm_lds(hipStream_t st,
    const _Float16* A, int lda, const _Float16* Bt, int ldb,
    const float* bias, float* Cf, _Float16* Ch, int ldc,
    int M, int N, int K, int gelu)
{
    dim3 grid(N / 64, M / 64, 1);
    gemm_wmma_lds<<<grid, 128, 0, st>>>(A, lda, Bt, ldb, bias, Cf, Ch, ldc,
                                        M, N, K, gelu);
}

extern "C" void kernel_launch(void* const* d_in, const int* in_sizes, int n_in,
                              void* d_out, int out_size, void* d_ws, size_t ws_size,
                              hipStream_t stream)
{
    (void)out_size; (void)ws_size; (void)n_in;
    // --- input mapping: insertion-order vs jax tree_flatten (sorted keys) ---
    const int *ids, *amask;
    const float *emb, *emb_ln_w, *emb_ln_b, *rel_emb, *rel_ln_w, *rel_ln_b;
    const float *Wq,*bq,*Wk,*bk,*Wv,*bv,*Wo,*bo,*ln1w,*ln1b,*Wi,*bi,*Wo2,*bo2,*ln2w,*ln2b;
    const float *preW,*preb,*cln1w,*cln1b,*cW1,*cb1,*cln2w,*cln2b,*cW2,*cb2;
#define FP(i) ((const float*)d_in[(i)])
    if ((long long)in_sizes[2] > 10000000LL) {
        // insertion order
        ids=(const int*)d_in[0]; amask=(const int*)d_in[1];
        emb=FP(2); emb_ln_w=FP(3); emb_ln_b=FP(4);
        rel_emb=FP(5); rel_ln_w=FP(6); rel_ln_b=FP(7);
        Wq=FP(8); bq=FP(9); Wk=FP(10); bk=FP(11); Wv=FP(12); bv=FP(13);
        Wo=FP(14); bo=FP(15); ln1w=FP(16); ln1b=FP(17);
        Wi=FP(18); bi=FP(19); Wo2=FP(20); bo2=FP(21); ln2w=FP(22); ln2b=FP(23);
        preW=FP(24); preb=FP(25); cln1w=FP(26); cln1b=FP(27);
        cW1=FP(28); cb1=FP(29); cln2w=FP(30); cln2b=FP(31); cW2=FP(32); cb2=FP(33);
    } else {
        // jax tree_flatten alphabetical order
        amask=(const int*)d_in[0]; ids=(const int*)d_in[1];
        cW1=FP(2); cW2=FP(3); cb1=FP(4); cb2=FP(5);
        cln1b=FP(6); cln1w=FP(7); cln2b=FP(8); cln2w=FP(9);
        emb=FP(10); emb_ln_b=FP(11); emb_ln_w=FP(12);
        Wi=FP(13); Wk=FP(14); Wo=FP(15); Wo2=FP(16); Wq=FP(17); Wv=FP(18);
        bi=FP(19); bk=FP(20); bo=FP(21); bo2=FP(22); bq=FP(23); bv=FP(24);
        ln1b=FP(25); ln1w=FP(26); ln2b=FP(27); ln2w=FP(28);
        preW=FP(29); preb=FP(30);
        rel_emb=FP(31); rel_ln_b=FP(32); rel_ln_w=FP(33);
    }
#undef FP

    // --- workspace layout (bump allocator, 256B aligned) ---
    char* base = (char*)d_ws;
    size_t off = 0;
    auto alloc = [&](size_t bytes) -> char* {
        off = (off + 255) & ~(size_t)255;
        char* p = base + off; off += bytes; return p;
    };
    float*     h32    = (float*)    alloc((size_t)MM * HH * 4);
    float*     tmp32  = (float*)    alloc((size_t)MM * HH * 4);
    _Float16*  h16    = (_Float16*) alloc((size_t)MM * HH * 2);
    _Float16*  q16    = (_Float16*) alloc((size_t)MM * HH * 2);
    _Float16*  k16    = (_Float16*) alloc((size_t)MM * HH * 2);
    _Float16*  v16    = (_Float16*) alloc((size_t)MM * HH * 2);
    _Float16*  ctx16  = (_Float16*) alloc((size_t)MM * HH * 2);
    _Float16*  rel16  = (_Float16*) alloc((size_t)SPAN2 * HH * 2);
    _Float16*  posk16 = (_Float16*) alloc((size_t)SPAN2 * HH * 2);
    _Float16*  posq16 = (_Float16*) alloc((size_t)SPAN2 * HH * 2);
    _Float16*  wq_t   = (_Float16*) alloc((size_t)HH * HH * 2);
    _Float16*  wk_t   = (_Float16*) alloc((size_t)HH * HH * 2);
    _Float16*  wv_t   = (_Float16*) alloc((size_t)HH * HH * 2);
    _Float16*  wo_t   = (_Float16*) alloc((size_t)HH * HH * 2);
    _Float16*  wi_t   = (_Float16*) alloc((size_t)HH * FFF * 2);
    _Float16*  wo2_t  = (_Float16*) alloc((size_t)HH * FFF * 2);
    _Float16*  ff1    = (_Float16*) alloc((size_t)MM * FFF * 2);
    _Float16*  vt     = (_Float16*) alloc((size_t)BB * NHH * DHH * SS * 2);
    float*     scores = (float*)    alloc((size_t)BB * NHH * SS * SS * 4);
    float*     cp     = (float*)    alloc((size_t)BB * NHH * SS * SS * 4);
    float*     pc     = (float*)    alloc((size_t)BB * NHH * SS * SS * 4);
    _Float16*  probs  = (_Float16*) alloc((size_t)BB * NHH * SS * SS * 2);
    int*       c2pI   = (int*)      alloc((size_t)SS * SS * 4);
    int*       p2cI   = (int*)      alloc((size_t)SS * SS * 4);
    float*     pooled = (float*)    alloc((size_t)BB * HH * 4);
    float*     clsA   = (float*)    alloc((size_t)BB * HH * 4);
    float*     clsB   = (float*)    alloc((size_t)BB * (HH/2) * 4);

    const float scale = 1.0f / sqrtf((float)(DHH * 3));  // 1/sqrt(192)

    // --- prologue: bucket tables, embeddings, relative embeddings ---
    bucket_kernel<<<(SS*SS + 255)/256, 256, 0, stream>>>(c2pI, p2cI);
    embed_ln<<<MM, 256, 0, stream>>>(ids, emb, emb_ln_w, emb_ln_b, h32, h16);
    ln768<<<SPAN2, 256, 0, stream>>>(rel_emb, nullptr, rel_ln_w, rel_ln_b, EPS_DB,
                                     nullptr, rel16);

    const long HL = (long)HH * HH;      // per-layer H*H weight stride
    const long SSz = (long)SS * SS;

    for (int l = 0; l < LL; ++l) {
        // weight prep (f32 [K][N] -> f16 [N][K])
        transpose_w<<<(int)((HL + 255)/256), 256, 0, stream>>>(Wq + (long)l*HL, wq_t, HH, HH);
        transpose_w<<<(int)((HL + 255)/256), 256, 0, stream>>>(Wk + (long)l*HL, wk_t, HH, HH);
        transpose_w<<<(int)((HL + 255)/256), 256, 0, stream>>>(Wv + (long)l*HL, wv_t, HH, HH);
        transpose_w<<<(int)((HL + 255)/256), 256, 0, stream>>>(Wo + (long)l*HL, wo_t, HH, HH);
        transpose_w<<<(int)(((long)HH*FFF + 255)/256), 256, 0, stream>>>(Wi  + (long)l*HH*FFF, wi_t,  HH,  FFF);
        transpose_w<<<(int)(((long)FFF*HH + 255)/256), 256, 0, stream>>>(Wo2 + (long)l*FFF*HH, wo2_t, FFF, HH);

        // Q/K/V projections (f16 out) — LDS-staged, async global->LDS
        launch_gemm_lds(stream, h16, HH, wq_t, HH, bq + l*HH, nullptr, q16, HH, MM, HH, HH, 0);
        launch_gemm_lds(stream, h16, HH, wk_t, HH, bk + l*HH, nullptr, k16, HH, MM, HH, HH, 0);
        launch_gemm_lds(stream, h16, HH, wv_t, HH, bv + l*HH, nullptr, v16, HH, MM, HH, HH, 0);
        // pos_k / pos_q projections of LN'd rel embeddings
        launch_gemm_lds(stream, rel16, HH, wk_t, HH, bk + l*HH, nullptr, posk16, HH, SPAN2, HH, HH, 0);
        launch_gemm_lds(stream, rel16, HH, wq_t, HH, bq + l*HH, nullptr, posq16, HH, SPAN2, HH, HH, 0);

        transpose_v<<<(BB*NHH*DHH*SS + 255)/256, 256, 0, stream>>>(v16, vt);

        // per-head batched GEMMs (z = b*NH + h, zInner = NH), direct-global
        // c2c: q·k^T
        launch_gemm(stream, q16, HH, (long)SS*HH, DHH,  k16, HH, (long)SS*HH, DHH,
                    nullptr, scores, nullptr, SS, (long)NHH*SSz, SSz,
                    SS, SS, DHH, BB*NHH, NHH, 0);
        // c2p: q·pos_k^T
        launch_gemm(stream, q16, HH, (long)SS*HH, DHH,  posk16, HH, 0, DHH,
                    nullptr, cp, nullptr, SS, (long)NHH*SSz, SSz,
                    SS, SPAN2, DHH, BB*NHH, NHH, 0);
        // p2c: k·pos_q^T
        launch_gemm(stream, k16, HH, (long)SS*HH, DHH,  posq16, HH, 0, DHH,
                    nullptr, pc, nullptr, SS, (long)NHH*SSz, SSz,
                    SS, SPAN2, DHH, BB*NHH, NHH, 0);

        combine_softmax<<<dim3(SS, BB*NHH), 512, 0, stream>>>(
            scores, cp, pc, c2pI, p2cI, amask, probs, scale);

        // ctx = probs · v   (B-operand = vt rows, K = S)
        launch_gemm(stream, probs, SS, (long)NHH*SSz, SSz,
                    vt, SS, (long)NHH*DHH*SS, (long)DHH*SS,
                    nullptr, nullptr, ctx16, HH, (long)SS*HH, DHH,
                    SS, DHH, SS, BB*NHH, NHH, 0);

        // output projection + residual + LN1
        launch_gemm_lds(stream, ctx16, HH, wo_t, HH, bo + l*HH, tmp32, nullptr, HH, MM, HH, HH, 0);
        ln768<<<MM, 256, 0, stream>>>(h32, tmp32, ln1w + l*HH, ln1b + l*HH, EPS_DB, h32, h16);

        // FFN: gelu(h·Wi + bi)·Wo2 + bo2, residual + LN2
        launch_gemm_lds(stream, h16, HH, wi_t, HH, bi + l*FFF, nullptr, ff1, FFF, MM, FFF, HH, 1);
        launch_gemm_lds(stream, ff1, FFF, wo2_t, FFF, bo2 + l*HH, tmp32, nullptr, HH, MM, HH, FFF, 0);
        ln768<<<MM, 256, 0, stream>>>(h32, tmp32, ln2w + l*HH, ln2b + l*HH, EPS_DB, h32, h16);
    }

    // --- masked mean pool + classifier head (tiny; f32) ---
    pool_kernel<<<BB, HH, 0, stream>>>(h32, amask, pooled);
    dense_small<<<(BB*HH + 127)/128, 128, 0, stream>>>(pooled, preW, preb, pooled, clsA, BB, HH, HH);
    ln_gelu_small<<<BB, 128, 0, stream>>>(clsA, cln1w, cln1b, HH, EPS_CLS);
    dense_small<<<(BB*(HH/2) + 127)/128, 128, 0, stream>>>(clsA, cW1, cb1, nullptr, clsB, BB, HH/2, HH);
    ln_gelu_small<<<BB, 128, 0, stream>>>(clsB, cln2w, cln2b, HH/2, EPS_CLS);
    dense_small<<<1, 128, 0, stream>>>(clsB, cW2, cb2, nullptr, (float*)d_out, BB, 2, HH/2);
}
